// DetectionLayer_48627619725485
// MI455X (gfx1250) — matrix operand order repaired and checked by reading, not compiled, not run on hardware
//
#include <hip/hip_runtime.h>
#include <stdint.h>

// ---------------- problem constants (match reference) ----------------
#define NROI      1000
#define NPAD      1024          // padded to power of two for bitonic sort
#define NCLS      81
#define MAX_INST  100
#define MIN_CONF  0.7f
#define NMS_T     0.3f
#define NEG_KEY   (-3.0e38f)    // sort key for invalid / padding entries

typedef __attribute__((ext_vector_type(16))) _Float16     v16h;
typedef __attribute__((ext_vector_type(8)))  float        v8f;
typedef __attribute__((ext_vector_type(4)))  unsigned int v4u;
typedef __attribute__((ext_vector_type(8)))  int          v8i;
typedef __attribute__((ext_vector_type(4)))  int          v4i;

__launch_bounds__(1024, 1)
__global__ void detection_layer_kernel(const float* __restrict__ rois,
                                       const float* __restrict__ probs,
                                       const float* __restrict__ deltas,
                                       const float* __restrict__ stddev,
                                       float* __restrict__ out) {
    // ---------------- LDS (~176 KB, fits CDNA5's 320 KB/WGP) ----------------
    __shared__ float        roistage[NROI * 4];     // 16000 B, filled by TDM
    __shared__ float        sby1[NPAD], sbx1[NPAD], sby2[NPAD], sbx2[NPAD]; // 16 KB
    __shared__ float        skey[NPAD];             // 4 KB  (sort key; == score for valid)
    __shared__ int          sidx[NPAD];             // 4 KB
    __shared__ int          scls[NPAD];             // 4 KB
    __shared__ unsigned int supw[NPAD * 32];        // 128 KB suppression bitmask
    __shared__ unsigned int keepw[32];              // final keep mask (1024 bits)
    __shared__ int          spfx[32];               // inclusive prefix popcounts (WMMA)
    __shared__ int          s_nv;

    const int tid = threadIdx.x;
    const int b   = blockIdx.x;

    // ---------------- TDM: async DMA rois[b] (1000x4 f32) -> LDS ----------------
    // D# built per CDNA5 ISA ch.8 (group0: count/lds/global/type; group1: dims).
    if (tid < 32) {
        uint64_t ga  = (uint64_t)(uintptr_t)(rois + (size_t)b * NROI * 4);
        uint32_t lds = (uint32_t)(uintptr_t)(&roistage[0]);   // flat LDS addr low 32 = LDS offset
        const uint32_t DIM0 = NROI * 4;                       // 4000 f32 elements
        v4u g0;
        g0[0] = 1u;                                           // count=1, user mode, no gather
        g0[1] = lds;                                          // lds_addr
        g0[2] = (uint32_t)(ga & 0xFFFFFFFFu);                 // global_addr[31:0]
        g0[3] = (uint32_t)((ga >> 32) & 0x01FFFFFFu) | (2u << 30); // addr[56:32] | type=2
        v8i g1;
        g1[0] = (int)(2u << 16);                              // wg_mask=0, data_size=2 (4B)
        g1[1] = (int)((DIM0 & 0xFFFFu) << 16);                // tensor_dim0[15:0] @ bits63:48
        g1[2] = (int)(((DIM0 >> 16) & 0xFFFFu) | (1u << 16)); // dim0 hi | tensor_dim1=1 lo
        g1[3] = (int)((DIM0 & 0xFFFFu) << 16);                // tensor_dim1 hi=0 | tile_dim0=4000
        g1[4] = (int)1u;                                      // tile_dim1=1, tile_dim2=0
        g1[5] = (int)DIM0;                                    // tensor_dim0_stride lo32
        g1[6] = 0; g1[7] = 0;                                 // stride hi, dim1_stride
        v4i g2 = {0, 0, 0, 0};
        v4i g3 = {0, 0, 0, 0};
        v8i g4 = {0, 0, 0, 0, 0, 0, 0, 0};                    // extra group (unused, <=2D tensor)
        __builtin_amdgcn_tensor_load_to_lds(g0, g1, g2, g3, g4, 0);
    }

    // ---------------- Phase A: class argmax + score (overlaps TDM) ----------------
    int   myCls = 0;
    float myScore = 0.0f;
    if (tid < NROI) {
        const float* p = probs + ((size_t)b * NROI + tid) * NCLS;
        float best = p[0];
        int   bc   = 0;
        for (int c = 1; c < NCLS; ++c) {
            float v = p[c];
            if (v > best) { best = v; bc = c; }   // strict > keeps first max (jnp.argmax)
        }
        myCls = bc; myScore = best;
    }
    const float sd0 = stddev[0], sd1 = stddev[1], sd2 = stddev[2], sd3 = stddev[3];

    if (tid < 32) __builtin_amdgcn_s_wait_tensorcnt(0);       // TDM done (wave 0)
    __syncthreads();                                          // LDS roistage visible to all

    // ---------------- Phase A2: decode, clip, stage into LDS ----------------
    if (tid < NROI) {
        float y1 = roistage[tid * 4 + 0], x1 = roistage[tid * 4 + 1];
        float y2 = roistage[tid * 4 + 2], x2 = roistage[tid * 4 + 3];
        // class-specific delta gather: 16B aligned float4
        const float4 d4 = *(const float4*)(deltas +
                           (((size_t)b * NROI + tid) * NCLS + myCls) * 4);
        float d0 = d4.x * sd0, d1 = d4.y * sd1, d2 = d4.z * sd2, d3 = d4.w * sd3;
        float h = y2 - y1, w = x2 - x1;
        float cy = y1 + 0.5f * h + d0 * h;
        float cx = x1 + 0.5f * w + d1 * w;
        h = h * expf(d2);
        w = w * expf(d3);
        float ny1 = cy - 0.5f * h, nx1 = cx - 0.5f * w;
        float ny2 = ny1 + h,       nx2 = nx1 + w;
        ny1 = fminf(fmaxf(ny1, 0.f), 1.f);
        nx1 = fminf(fmaxf(nx1, 0.f), 1.f);
        ny2 = fminf(fmaxf(ny2, 0.f), 1.f);
        nx2 = fminf(fmaxf(nx2, 0.f), 1.f);
        bool valid = (myCls > 0) && (myScore >= MIN_CONF);
        sby1[tid] = ny1; sbx1[tid] = nx1; sby2[tid] = ny2; sbx2[tid] = nx2;
        skey[tid] = valid ? myScore : NEG_KEY;
        scls[tid] = myCls;
        sidx[tid] = tid;
    } else {                                  // padding entries 1000..1023
        sby1[tid] = 0.f; sbx1[tid] = 0.f; sby2[tid] = 0.f; sbx2[tid] = 0.f;
        skey[tid] = NEG_KEY;
        scls[tid] = 0;
        sidx[tid] = tid;
    }
    __syncthreads();

    // ---------------- Phase B: bitonic sort 1024 (key desc, idx asc = stable) ----------------
    for (int k = 2; k <= NPAD; k <<= 1) {
        for (int j = k >> 1; j > 0; j >>= 1) {
            int ixj = tid ^ j;
            if (ixj > tid) {
                float ka = skey[tid], kb = skey[ixj];
                int   ia = sidx[tid], ib = sidx[ixj];
                bool bBeforeA  = (kb > ka) || ((kb == ka) && (ib < ia));
                bool descBlock = ((tid & k) == 0);
                if (descBlock ? bBeforeA : !bBeforeA) {
                    skey[tid] = kb; skey[ixj] = ka;
                    sidx[tid] = ib; sidx[ixj] = ia;
                }
            }
            __syncthreads();
        }
    }

    // ---------------- Phase B2: gather payloads into sorted order ----------------
    {
        int src = sidx[tid];
        float gy1 = sby1[src], gx1 = sbx1[src], gy2 = sby2[src], gx2 = sbx2[src];
        int   gc  = scls[src];
        __syncthreads();
        sby1[tid] = gy1; sbx1[tid] = gx1; sby2[tid] = gy2; sbx2[tid] = gx2;
        scls[tid] = gc;
    }
    __syncthreads();

    // ---------------- Phase C: count valid (sorted => contiguous prefix) ----------------
    {
        bool vme = skey[tid] > -1.0e30f;
        bool vnx = (tid < NPAD - 1) ? (skey[tid + 1] > -1.0e30f) : false;
        if (tid == 0 && !vme) s_nv = 0;
        if (vme && !vnx)      s_nv = tid + 1;
    }
    __syncthreads();
    const int nv = s_nv;

    // ---------------- Phase D: suppression bitmask rows [0, nv) ----------------
    for (int w = tid; w < (nv << 5); w += (int)blockDim.x) {
        int   i   = w >> 5;
        int   wo  = w & 31;
        float iy1 = sby1[i], ix1 = sbx1[i], iy2 = sby2[i], ix2 = sbx2[i];
        float ia  = (iy2 - iy1) * (ix2 - ix1);
        int   ic  = scls[i];
        unsigned int bits = 0u;
        int jb = wo << 5;
        for (int t = 0; t < 32; ++t) {
            int   j   = jb + t;
            float jy1 = sby1[j], jx1 = sbx1[j], jy2 = sby2[j], jx2 = sbx2[j];
            float iy  = fmaxf(fminf(iy2, jy2) - fmaxf(iy1, jy1), 0.f);
            float ix  = fmaxf(fminf(ix2, jx2) - fmaxf(ix1, jx1), 0.f);
            float inter = iy * ix;
            float ja  = (jy2 - jy1) * (jx2 - jx1);
            float uni = fmaxf(ia + ja - inter, 1e-8f);
            bool  sup = ((inter / uni) > NMS_T) && (ic == scls[j]);
            bits |= (sup ? 1u : 0u) << t;
        }
        supw[(i << 5) | wo] = bits;
    }
    __syncthreads();

    // ---------------- Phase E: greedy NMS scan, single wave, bitmask in registers ----------------
    if (tid < 32) {
        int base = tid << 5;
        unsigned int keep;
        if (nv >= base + 32)      keep = 0xFFFFFFFFu;   // all valid
        else if (nv <= base)      keep = 0u;
        else                      keep = (1u << (nv - base)) - 1u;
        for (int i = 0; i < nv; ++i) {
            unsigned int wk = (unsigned int)__shfl((int)keep, i >> 5, 32);
            if ((wk >> (i & 31)) & 1u) {
                unsigned int row = supw[(i << 5) | tid];
                int iw = i >> 5, ib = i & 31;
                unsigned int m;
                if (tid < iw)       m = 0u;
                else if (tid > iw)  m = 0xFFFFFFFFu;
                else                m = (ib == 31) ? 0u : (0xFFFFFFFFu << (ib + 1));
                keep &= ~(row & m);   // suppress only j > i
            }
        }
        keepw[tid] = keep;
    }
    __syncthreads();

    // ---------------- Phase F: WMMA prefix-sum of per-word popcounts ----------------
    // p[m] = sum_{k<=m} v[k] via D = L * broadcast(v); exact in f16/f32 (v<=32, sums<=1024).
    {
        int lane = tid & 31;
        int m    = lane & 15;
        v16h A1, A2, Bv;
        for (int e = 0; e < 16; ++e) {
            // A 16x32 f16 layout: lanes0-15 e<8->K=e, e>=8->K=e+8; lanes16-31 e<8->K=e+8, else K=e+16
            int Ka = (lane < 16) ? ((e < 8) ? e : e + 8) : ((e < 8) ? e + 8 : e + 16);
            // B 32x16 f16 layout: same K mapping as A
            int Kb = Ka;
            A1[e] = (_Float16)((Ka <= m)      ? 1.0f : 0.0f);
            A2[e] = (_Float16)((Ka <= m + 16) ? 1.0f : 0.0f);
            Bv[e] = (_Float16)(float)__popc(keepw[Kb]);
        }
        v8f Cz = {0.f, 0.f, 0.f, 0.f, 0.f, 0.f, 0.f, 0.f};
        v8f D1 = __builtin_amdgcn_wmma_f32_16x16x32_f16(false, A1, false, Bv,
                                                        (short)0, Cz, false, false);
        v8f D2 = __builtin_amdgcn_wmma_f32_16x16x32_f16(false, A2, false, Bv,
                                                        (short)0, Cz, false, false);
        // D layout: lanes0-15 elem r -> M=r; lanes16-31 elem r -> M=8+r
        if (tid == 0) {
            for (int r = 0; r < 8; ++r) {
                spfx[r]      = (int)(D1[r] + 0.5f);
                spfx[16 + r] = (int)(D2[r] + 0.5f);
            }
        }
        if (tid == 16) {
            for (int r = 0; r < 8; ++r) {
                spfx[8 + r]  = (int)(D1[r] + 0.5f);
                spfx[24 + r] = (int)(D2[r] + 0.5f);
            }
        }
    }
    __syncthreads();

    // ---------------- Phase G: zero output, write ranked rows ----------------
    float* outb = out + (size_t)b * MAX_INST * 6;
    for (int t = tid; t < MAX_INST * 6; t += (int)blockDim.x) outb[t] = 0.0f;
    __syncthreads();
    {
        int wj = tid >> 5, bj = tid & 31;
        unsigned int kw = keepw[wj];
        if ((kw >> bj) & 1u) {
            int rank = ((wj > 0) ? spfx[wj - 1] : 0) +
                       __popc(kw & ((bj == 0) ? 0u : ((1u << bj) - 1u)));
            if (rank < MAX_INST) {
                float* row = outb + rank * 6;
                row[0] = sby1[tid]; row[1] = sbx1[tid];
                row[2] = sby2[tid]; row[3] = sbx2[tid];
                row[4] = (float)scls[tid];
                row[5] = skey[tid];           // kept => valid => key == score
            }
        }
    }
}

extern "C" void kernel_launch(void* const* d_in, const int* in_sizes, int n_in,
                              void* d_out, int out_size, void* d_ws, size_t ws_size,
                              hipStream_t stream) {
    const float* rois   = (const float*)d_in[0];
    const float* probs  = (const float*)d_in[1];
    const float* deltas = (const float*)d_in[2];
    const float* stddev = (const float*)d_in[3];
    float* out = (float*)d_out;
    (void)d_ws; (void)ws_size; (void)n_in; (void)out_size;
    int batches = in_sizes[0] / (NROI * 4);   // 32
    detection_layer_kernel<<<dim3(batches), dim3(1024), 0, stream>>>(
        rois, probs, deltas, stddev, out);
}
